// GNN_83279415870039
// MI455X (gfx1250) — matrix operand order repaired and checked by reading, not compiled, not run on hardware
//
#include <hip/hip_runtime.h>
#include <hip/hip_bf16.h>
#include <math.h>

typedef _Float16 f16;
typedef __attribute__((ext_vector_type(16))) _Float16 v16h;
typedef __attribute__((ext_vector_type(8)))  float    v8f;
typedef __attribute__((ext_vector_type(4)))  unsigned u32x4;
typedef __attribute__((ext_vector_type(8)))  int      i32x8;
typedef __attribute__((ext_vector_type(4)))  int      i32x4;

#ifndef __has_builtin
#define __has_builtin(x) 0
#endif
#if __has_builtin(__builtin_amdgcn_tensor_load_to_lds) && __has_builtin(__builtin_amdgcn_s_wait_tensorcnt)
#define USE_TDM 1
#else
#define USE_TDM 0
#endif

#define NN      32768      // nodes
#define NE      262144     // edges
#define NG      8          // graphs
#define NPER    4096
#define CH      64
#define LAYERS  6
#define KPOOL   16
#define HLSTM   192
#define E2T     (NE + NN)  // edges + self loops

// padded LDS addressing: insert 8 f16 after every 64 f16 (4 dwords per 32 dwords)
constexpr int pad64(int li) { return li + ((li >> 6) << 3); }

// ---------------------------------------------------------------- WMMA core
__device__ __forceinline__ v8f wmma_f16(v16h a, v16h b, v8f c) {
  return __builtin_amdgcn_wmma_f32_16x16x32_f16(false, a, false, b, (short)0, c, false, false);
}

// A fragment from an LDS tile (16 x K row-major, leading dim lda) -> two ds_load_b128
__device__ __forceinline__ v16h frag_a_lds(const f16* A, int lda, int kBase) {
  const int lane = threadIdx.x;
  const f16* p = A + (lane & 15) * lda + kBase + ((lane >> 4) & 1) * 8;
  v16h a;
  ((uint4*)&a)[0] = ((const uint4*)p)[0];
  ((uint4*)&a)[1] = ((const uint4*)(p + 16))[0];
  return a;
}
// A fragment from a pad64-laid-out LDS tile (16 x K logical)
template <int K>
__device__ __forceinline__ v16h frag_a_pad(const f16* As, int kBase) {
  const int lane = threadIdx.x;
  const int li0 = (lane & 15) * K + kBase + ((lane >> 4) & 1) * 8;
  const int li1 = li0 + 16;
  v16h a;
  ((uint4*)&a)[0] = ((const uint4*)(As + pad64(li0)))[0];
  ((uint4*)&a)[1] = ((const uint4*)(As + pad64(li1)))[0];
  return a;
}
// B fragment from fragment-packed weights: ((kt*nCT + ct)*32 + lane)*16 contiguous f16
__device__ __forceinline__ v16h frag_b_packed(const f16* Bp, int kt, int nCT, int ct) {
  const f16* p = Bp + ((((size_t)kt * nCT + ct) * 32 + threadIdx.x) << 4);
  v16h b;
  ((uint4*)&b)[0] = ((const uint4*)p)[0];
  ((uint4*)&b)[1] = ((const uint4*)p)[1];
  return b;
}

// ---------------------------------------------------------------- utilities
__device__ __forceinline__ void atomicMaxF(float* addr, float val) {
  unsigned* ai = (unsigned*)addr;
  unsigned old = *ai;
  while (true) {
    float f = __uint_as_float(old);
    if (f >= val) break;
    unsigned assumed = old;
    old = atomicCAS(ai, assumed, __float_as_uint(val));
    if (old == assumed) break;
  }
}
__device__ __forceinline__ void atomicMinF(float* addr, float val) {
  unsigned* ai = (unsigned*)addr;
  unsigned old = *ai;
  while (true) {
    float f = __uint_as_float(old);
    if (f <= val) break;
    unsigned assumed = old;
    old = atomicCAS(ai, assumed, __float_as_uint(val));
    if (old == assumed) break;
  }
}

__global__ void k_fill_f32(float* p, float v, int n) {
  int i = blockIdx.x * blockDim.x + threadIdx.x;
  if (i < n) p[i] = v;
}
__global__ void k_zero_u32(unsigned* p, int n) {
  int i = blockIdx.x * blockDim.x + threadIdx.x;
  if (i < n) p[i] = 0u;
}
__global__ void k_copy_f32(float* d, const float* s, int n) {
  int i = blockIdx.x * blockDim.x + threadIdx.x;
  if (i < n) d[i] = s[i];
}
__global__ void k_f32_to_f16(f16* d, const float* s, int n) {
  int i = blockIdx.x * blockDim.x + threadIdx.x;
  if (i < n) d[i] = (f16)s[i];
}
// weight [Co,K] f32 -> fragment-packed f16 for frag_b_packed()
__global__ void k_wFrag(f16* dst, const float* src, int Co, int K) {
  int i = blockIdx.x * blockDim.x + threadIdx.x;
  if (i >= K * Co) return;
  int e = i & 15;
  int lane = (i >> 4) & 31;
  int rest = i >> 9;                // kt*nCT + ct
  int nCT = Co >> 4;
  int kt = rest / nCT, ct = rest - kt * nCT;
  int n = lane & 15, kh = (lane >> 4) & 1;
  int k = kt * 32 + (e < 8 ? kh * 8 + e : 16 + kh * 8 + (e - 8));
  int co = ct * 16 + n;
  dst[i] = (f16)src[(size_t)co * K + k];
}

// ---------------------------------------------------------------- PNA layer
__global__ void k_edge_e(const float* __restrict__ ea, const float* __restrict__ ew,
                         const float* __restrict__ eb, f16* __restrict__ ef) {
  int idx = blockIdx.x * blockDim.x + threadIdx.x;  // e*64 + c
  if (idx >= NE * CH) return;
  int e = idx >> 6, c = idx & 63;
  const float* a = ea + (size_t)e * 8;
  const float* w = ew + (size_t)c * 8;
  float s = eb[c];
#pragma unroll
  for (int k = 0; k < 8; ++k) s += a[k] * w[k];
  ef[idx] = (f16)s;
}

// m[16 edges x 64] = gather([x_dst | x_src | e]) @ preT + pre_b ; WMMA K=192
__global__ void __launch_bounds__(128)
k_msg_gemm(const f16* __restrict__ xcur, const int* __restrict__ src, const int* __restrict__ dst,
           const f16* __restrict__ ef, const f16* __restrict__ preT,
           const float* __restrict__ preB, float* __restrict__ m) {
  __shared__ f16 As[16 * 200];
  const int e0 = blockIdx.x * 16;
  const int tid = threadIdx.y * 32 + threadIdx.x;
  for (int ci = tid; ci < 16 * 24; ci += 128) {  // 16 rows x 24 chunks of 8 f16
    int r = ci / 24, ch = ci - r * 24;
    int col = ch * 8;
    int ge = e0 + r;
    const f16* sp;
    if (col < 64)       sp = xcur + (size_t)dst[ge] * CH + col;
    else if (col < 128) sp = xcur + (size_t)src[ge] * CH + (col - 64);
    else                sp = ef + (size_t)ge * CH + (col - 128);
    *(uint4*)(As + r * 200 + col) = *(const uint4*)sp;
  }
  __syncthreads();
  const int ct = threadIdx.y;  // column tile, nCT = 4
  v8f acc = {};
#pragma unroll
  for (int kt = 0; kt < 6; ++kt)
    acc = wmma_f16(frag_a_lds(As, 200, kt * 32), frag_b_packed(preT, kt, 4, ct), acc);
  const int lane = threadIdx.x;
  const int n = lane & 15, mb = (lane >> 4) * 8;
  float bias = preB[ct * 16 + n];
#pragma unroll
  for (int r = 0; r < 8; ++r)
    m[(size_t)(e0 + mb + r) * CH + ct * 16 + n] = acc[r] + bias;
}

__global__ void k_scatter_agg(const int* __restrict__ dst, const float* __restrict__ m,
                              float* s1, float* s2, float* mx, float* mn, float* deg) {
  int idx = blockIdx.x * blockDim.x + threadIdx.x;  // e*64+c
  if (idx >= NE * CH) return;
  int e = idx >> 6, c = idx & 63;
  int d = dst[e];
  float v = m[idx];
  atomicAdd(&s1[(size_t)d * CH + c], v);
  atomicAdd(&s2[(size_t)d * CH + c], v * v);
  atomicMaxF(&mx[(size_t)d * CH + c], v);
  atomicMinF(&mn[(size_t)d * CH + c], v);
  if (c == 0) atomicAdd(&deg[d], 1.f);
}

__global__ void k_build_out9(const float* __restrict__ x, const float* __restrict__ s1,
                             const float* __restrict__ s2, const float* __restrict__ mx,
                             const float* __restrict__ mn, const float* __restrict__ deg,
                             const float* __restrict__ adl, f16* __restrict__ out9) {
  int idx = blockIdx.x * blockDim.x + threadIdx.x;  // n*64+c
  if (idx >= NN * CH) return;
  int n = idx >> 6, c = idx & 63;
  float dg = deg[n];
  float degc = fmaxf(dg, 1.f);
  float mean = s1[idx] / degc;
  float var = s2[idx] / degc - mean * mean;
  float stdv = sqrtf(fmaxf(var, 0.f) + 1e-5f);
  bool has = dg > 0.f;
  float mxv = has ? mx[idx] : 0.f;
  float mnv = has ? mn[idx] : 0.f;
  float amp = logf(degc + 1.f) / adl[0];
  f16* row = out9 + (size_t)n * 576;
  row[c]       = (f16)x[idx];
  row[64 + c]  = (f16)mean;
  row[128 + c] = (f16)stdv;
  row[192 + c] = (f16)mxv;
  row[256 + c] = (f16)mnv;
  row[320 + c] = (f16)(mean * amp);
  row[384 + c] = (f16)(stdv * amp);
  row[448 + c] = (f16)(mxv * amp);
  row[512 + c] = (f16)(mnv * amp);
}

// generic WMMA GEMM: C[M,64] = A[M,K](f16) @ Bfrag + bias ; K compile-time, M%16==0
// A tile staged via the Tensor Data Mover (TDM pads LDS rows itself) when available.
template <int K>
__global__ void __launch_bounds__(128)
k_gemm64(const f16* __restrict__ A, const f16* __restrict__ Bp,
         const float* __restrict__ bias, float* __restrict__ outF, f16* __restrict__ outH) {
  __shared__ f16 As[pad64(16 * K)];
  const int row0 = blockIdx.x * 16;
  const int tid = threadIdx.y * 32 + threadIdx.x;
  (void)tid;
#if USE_TDM
  if (threadIdx.y == 0) {
    // Tensor DMA descriptor: 2D tile 16 x K f16, global row stride K, LDS pad
    // 8 f16 per 64 f16 (pad_interval=4 -> 128B groups, pad_amount=3 -> 4 dwords).
    unsigned lds = (unsigned)(size_t)(&As[0]);
    unsigned long long ga = (unsigned long long)(size_t)(A + (size_t)row0 * K);
    u32x4 g0;
    g0[0] = 1u;                                            // count=1, user D#
    g0[1] = lds;                                           // lds_addr
    g0[2] = (unsigned)(ga & 0xffffffffu);                  // global_addr lo
    g0[3] = (unsigned)((ga >> 32) & 0x01ffffffu) | (2u << 30);  // addr hi | type=2
    i32x8 g1;
    g1[0] = (int)((1u << 16) | (1u << 20) | (4u << 22) | (3u << 25));  // 2B, pad on
    g1[1] = (int)((unsigned)(K & 0xffff) << 16);           // tensor_dim0 lo
    g1[2] = (int)(16u << 16);                              // tensor_dim1 = 16
    g1[3] = (int)((unsigned)(K & 0xffff) << 16);           // tile_dim0 = K
    g1[4] = 16;                                            // tile_dim1 = 16
    g1[5] = K;                                             // dim0 stride lo
    g1[6] = 0;
    g1[7] = 0;
    i32x4 g2 = {0, 0, 0, 0};
    i32x4 g3 = {0, 0, 0, 0};
#if defined(__clang_major__) && (__clang_major__ >= 23)
    i32x8 g4 = {0, 0, 0, 0, 0, 0, 0, 0};
    __builtin_amdgcn_tensor_load_to_lds(g0, g1, g2, g3, g4, 0);
#else
    __builtin_amdgcn_tensor_load_to_lds(g0, g1, g2, g3, 0);
#endif
    __builtin_amdgcn_s_wait_tensorcnt((short)0);
  }
#else
  constexpr int RCH = K / 8;  // 16B chunks per row
  for (int ci = tid; ci < 16 * RCH; ci += 128) {
    int r = ci / RCH, ch = ci - r * RCH;
    int col = ch * 8;
    int li = r * K + col;
    *(uint4*)(As + pad64(li)) = *(const uint4*)(A + (size_t)(row0 + r) * K + col);
  }
#endif
  __builtin_prefetch(A + (size_t)(row0 + 16) * K, 0, 1);  // next row-tile
  __syncthreads();
  const int ct = threadIdx.y;
  v8f acc = {};
#pragma unroll
  for (int kt = 0; kt < K / 32; ++kt)
    acc = wmma_f16(frag_a_pad<K>(As, kt * 32), frag_b_packed(Bp, kt, 4, ct), acc);
  const int lane = threadIdx.x;
  const int n = lane & 15, mb = (lane >> 4) * 8;
  float bv = bias ? bias[ct * 16 + n] : 0.f;
#pragma unroll
  for (int r = 0; r < 8; ++r) {
    int row = row0 + mb + r;
    float v = acc[r] + bv;
    if (outF) outF[(size_t)row * CH + ct * 16 + n] = v;
    if (outH) outH[(size_t)row * CH + ct * 16 + n] = (f16)v;
  }
}

// ---------------------------------------------------------------- JK bi-LSTM
__global__ void __launch_bounds__(128)
k_lstm_step(const f16* __restrict__ xt, const f16* __restrict__ hIn, f16* __restrict__ hOut,
            float* __restrict__ cSt, const f16* __restrict__ wihP, const f16* __restrict__ whhP,
            const float* __restrict__ bih, const float* __restrict__ bhh,
            const float* __restrict__ attw, int attOff, float* __restrict__ alphaT) {
  __shared__ f16 Ax[16 * 72];
  __shared__ f16 Ah[16 * 200];
  __shared__ float G[4][16][16];
  const int row0 = blockIdx.x * 16;
  const int jt = blockIdx.y;
  const int tid = threadIdx.y * 32 + threadIdx.x;
  for (int ci = tid; ci < 16 * 8; ci += 128) {  // 16 rows x 8 chunks (64 f16)
    int r = ci >> 3, col = (ci & 7) * 8;
    *(uint4*)(Ax + r * 72 + col) = *(const uint4*)(xt + (size_t)(row0 + r) * CH + col);
  }
  for (int ci = tid; ci < 16 * 24; ci += 128) {  // 16 rows x 24 chunks (192 f16)
    int r = ci / 24, col = (ci - r * 24) * 8;
    *(uint4*)(Ah + r * 200 + col) = *(const uint4*)(hIn + (size_t)(row0 + r) * HLSTM + col);
  }
  __syncthreads();
  const int gate = threadIdx.y;
  const int ct = gate * 12 + jt;  // column tile in the 768-wide gate matrix (nCT = 48)
  v8f acc = {};
#pragma unroll
  for (int kt = 0; kt < 2; ++kt)
    acc = wmma_f16(frag_a_lds(Ax, 72, kt * 32), frag_b_packed(wihP, kt, 48, ct), acc);
#pragma unroll
  for (int kt = 0; kt < 6; ++kt)
    acc = wmma_f16(frag_a_lds(Ah, 200, kt * 32), frag_b_packed(whhP, kt, 48, ct), acc);
  const int lane = threadIdx.x;
  const int n = lane & 15, mb = (lane >> 4) * 8;
  const int col0 = ct * 16;
  float bb = bih[col0 + n] + bhh[col0 + n];
#pragma unroll
  for (int r = 0; r < 8; ++r) G[gate][mb + r][n] = acc[r] + bb;
  __syncthreads();
#pragma unroll
  for (int q = 0; q < 2; ++q) {
    int idx = tid * 2 + q;  // 0..255 = 16 rows x 16 units
    int r = idx >> 4, u = idx & 15;
    int node = row0 + r, unit = jt * 16 + u;
    float gi = G[0][r][u], gf = G[1][r][u], gg = G[2][r][u], go = G[3][r][u];
    float si = 1.f / (1.f + expf(-gi));
    float sf = 1.f / (1.f + expf(-gf));
    float so = 1.f / (1.f + expf(-go));
    float cOld = cSt[(size_t)node * HLSTM + unit];
    float cNew = sf * cOld + si * tanhf(gg);
    float hNew = so * tanhf(cNew);
    cSt[(size_t)node * HLSTM + unit] = cNew;
    hOut[(size_t)node * HLSTM + unit] = (f16)hNew;
    atomicAdd(&alphaT[node], attw[attOff + unit] * hNew);
  }
}

__global__ void k_jk_combine(const float* __restrict__ alpha, const float* __restrict__ attb,
                             const float* __restrict__ xs, float* __restrict__ xo) {
  int idx = blockIdx.x * blockDim.x + threadIdx.x;  // n*64+c
  if (idx >= NN * CH) return;
  int n = idx >> 6;
  float a[LAYERS], mxv = -1e30f;
#pragma unroll
  for (int t = 0; t < LAYERS; ++t) { a[t] = alpha[t * NN + n] + attb[0]; mxv = fmaxf(mxv, a[t]); }
  float s = 0.f;
#pragma unroll
  for (int t = 0; t < LAYERS; ++t) { a[t] = expf(a[t] - mxv); s += a[t]; }
  float out = 0.f;
#pragma unroll
  for (int t = 0; t < LAYERS; ++t) out += (a[t] / s) * xs[(size_t)t * NN * CH + idx];
  xo[idx] = out;
}

// ---------------------------------------------------------------- ASAP pooling
__global__ void k_scatter_max_x(const int* __restrict__ src, const int* __restrict__ dst,
                                const float* __restrict__ x, float* mx) {
  int idx = blockIdx.x * blockDim.x + threadIdx.x;  // e*64+c (real edges only)
  if (idx >= NE * CH) return;
  int e = idx >> 6, c = idx & 63;
  atomicMaxF(&mx[(size_t)dst[e] * CH + c], x[(size_t)src[e] * CH + c]);
}

__global__ void k_asap_score(const int* __restrict__ src, const int* __restrict__ dst,
                             const float* __restrict__ xq, const float* __restrict__ x,
                             const float* __restrict__ attw, const float* __restrict__ attb,
                             float* __restrict__ s) {
  int e = blockIdx.x * blockDim.x + threadIdx.x;
  if (e >= E2T) return;
  int ss = e < NE ? src[e] : (e - NE);
  int dd = e < NE ? dst[e] : (e - NE);
  float acc = attb[0];
  for (int c = 0; c < CH; ++c) acc += attw[c] * xq[(size_t)dd * CH + c];
  for (int c = 0; c < CH; ++c) acc += attw[CH + c] * x[(size_t)ss * CH + c];
  s[e] = acc > 0.f ? acc : 0.2f * acc;  // leaky_relu(0.2)
}
__global__ void k_scatter_smax(const int* __restrict__ dst, const float* __restrict__ s, float* smax) {
  int e = blockIdx.x * blockDim.x + threadIdx.x;
  if (e >= E2T) return;
  int dd = e < NE ? dst[e] : (e - NE);
  atomicMaxF(&smax[dd], s[e]);
}
__global__ void k_seg_exp(const int* __restrict__ dst, const float* __restrict__ s,
                          const float* __restrict__ smax, float* es, float* ssum) {
  int e = blockIdx.x * blockDim.x + threadIdx.x;
  if (e >= E2T) return;
  int dd = e < NE ? dst[e] : (e - NE);
  float v = expf(s[e] - smax[dd]);
  es[e] = v;
  atomicAdd(&ssum[dd], v);
}
__global__ void k_scatter_v(const int* __restrict__ src, const int* __restrict__ dst,
                            const float* __restrict__ x, const float* __restrict__ es,
                            const float* __restrict__ ssum, float* v) {
  int idx = blockIdx.x * blockDim.x + threadIdx.x;  // e*64+c
  if (idx >= E2T * CH) return;
  int e = idx >> 6, c = idx & 63;
  int ss = e < NE ? src[e] : (e - NE);
  int dd = e < NE ? dst[e] : (e - NE);
  float score = es[e] / (ssum[dd] + 1e-16f);
  atomicAdd(&v[(size_t)dd * CH + c], x[(size_t)ss * CH + c] * score);
}
__global__ void k_leconv_node(const float* __restrict__ v, const float* __restrict__ w1,
                              const float* __restrict__ b1, const float* __restrict__ w2,
                              const float* __restrict__ w3, const float* __restrict__ b3,
                              float* aN, float* bN, float* fit) {
  int n = blockIdx.x * blockDim.x + threadIdx.x;
  if (n >= NN) return;
  float a = 0.f, b = 0.f, c3 = 0.f;
  for (int c = 0; c < CH; ++c) {
    float x = v[(size_t)n * CH + c];
    a += x * w1[c]; b += x * w2[c]; c3 += x * w3[c];
  }
  aN[n] = a + b1[0]; bN[n] = b; fit[n] = c3 + b3[0];
}
__global__ void k_scatter_fit(const int* __restrict__ src, const int* __restrict__ dst,
                              const float* __restrict__ aN, const float* __restrict__ bN,
                              float* fit) {
  int e = blockIdx.x * blockDim.x + threadIdx.x;
  if (e >= E2T) return;
  int ss = e < NE ? src[e] : (e - NE);
  int dd = e < NE ? dst[e] : (e - NE);
  atomicAdd(&fit[dd], aN[ss] - bN[dd]);
}
__global__ void k_sigmoid(const float* in, float* out, int n) {
  int i = blockIdx.x * blockDim.x + threadIdx.x;
  if (i < n) out[i] = 1.f / (1.f + expf(-in[i]));
}

// per-graph iterative top-16 + gather of v[perm]*fitness[perm] into the MLP input
__global__ void __launch_bounds__(256)
k_topk_gather(const float* __restrict__ fitness, const float* __restrict__ v,
              float* __restrict__ mlpin) {
  __shared__ float bv[256];
  __shared__ int bi[256];
  __shared__ int sel[KPOOL];
  const int g = blockIdx.x;
  const int tid = threadIdx.x;
  for (int k = 0; k < KPOOL; ++k) {
    float best = -1e30f; int bidx = 0x7fffffff;
    for (int nl = tid; nl < NPER; nl += 256) {
      bool taken = false;
      for (int j = 0; j < k; ++j) if (sel[j] == nl) taken = true;
      if (taken) continue;
      float f = fitness[g * NPER + nl];
      if (f > best || (f == best && nl < bidx)) { best = f; bidx = nl; }
    }
    bv[tid] = best; bi[tid] = bidx;
    __syncthreads();
    for (int off = 128; off > 0; off >>= 1) {
      if (tid < off) {
        float fo = bv[tid + off]; int io = bi[tid + off];
        if (fo > bv[tid] || (fo == bv[tid] && io < bi[tid])) { bv[tid] = fo; bi[tid] = io; }
      }
      __syncthreads();
    }
    if (tid == 0) sel[k] = bi[0];
    __syncthreads();
    if (tid < CH) {
      int node = g * NPER + sel[k];
      float val = v[(size_t)node * CH + tid] * fitness[node];
      if (isnan(val)) val = 0.f;
      else if (isinf(val)) val = val > 0.f ? 3.4028235e38f : -3.4028235e38f;
      mlpin[g * (KPOOL * CH) + k * CH + tid] = val;
    }
    __syncthreads();
  }
}

// tiny dense layer (8 rows) with exact GELU
__global__ void k_dense(const float* __restrict__ in, const float* __restrict__ w,
                        const float* __restrict__ b, float* __restrict__ out,
                        int K, int Co, int act) {
  int idx = blockIdx.x * blockDim.x + threadIdx.x;
  if (idx >= NG * Co) return;
  int row = idx / Co, co = idx - row * Co;
  float acc = b[co];
  for (int k = 0; k < K; ++k) acc += in[(size_t)row * K + k] * w[(size_t)co * K + k];
  if (act) acc = 0.5f * acc * (1.f + erff(acc * 0.70710678118654752f));
  out[idx] = acc;
}

// ---------------------------------------------------------------- host
static inline unsigned gdiv(long n, int b) { return (unsigned)((n + b - 1) / b); }
static inline size_t alignUp(size_t x) { return (x + 255) & ~(size_t)255; }

extern "C" void kernel_launch(void* const* d_in, const int* in_sizes, int n_in,
                              void* d_out, int out_size, void* d_ws, size_t ws_size,
                              hipStream_t stream) {
  (void)in_sizes; (void)n_in; (void)out_size; (void)ws_size;
  const float* x_in   = (const float*)d_in[0];
  const float* ea_in  = (const float*)d_in[1];
  const int*   ei     = (const int*)d_in[2];
  const int*   srcA   = ei;
  const int*   dstA   = ei + NE;
  const float* edge_w = (const float*)d_in[4];
  const float* edge_b = (const float*)d_in[5];
  const float* pre_w  = (const float*)d_in[6];
  const float* pre_b  = (const float*)d_in[7];
  const float* post_w = (const float*)d_in[8];
  const float* post_b = (const float*)d_in[9];
  const float* lin_w  = (const float*)d_in[10];
  const float* lin_b  = (const float*)d_in[11];
  const float* adl    = (const float*)d_in[12];
  const float* wihF   = (const float*)d_in[13];
  const float* whhF   = (const float*)d_in[14];
  const float* bihF   = (const float*)d_in[15];
  const float* bhhF   = (const float*)d_in[16];
  const float* wihB   = (const float*)d_in[17];
  const float* whhB   = (const float*)d_in[18];
  const float* bihB   = (const float*)d_in[19];
  const float* bhhB   = (const float*)d_in[20];
  const float* jkAttW = (const float*)d_in[21];
  const float* jkAttB = (const float*)d_in[22];
  const float* asLinW = (const float*)d_in[23];
  const float* asLinB = (const float*)d_in[24];
  const float* asAttW = (const float*)d_in[25];
  const float* asAttB = (const float*)d_in[26];
  const float* leW1   = (const float*)d_in[27];
  const float* leB1   = (const float*)d_in[28];
  const float* leW2   = (const float*)d_in[29];
  const float* leW3   = (const float*)d_in[30];
  const float* leB3   = (const float*)d_in[31];
  float* out = (float*)d_out;

  // -------- workspace layout (phase1 / phase2 regions overlap in memory, not in time)
  char* ws = (char*)d_ws;
  size_t off = 0;
  auto alloc = [&](size_t bytes) -> char* { char* p = ws + off; off = alignUp(off + bytes); return p; };
  float* xs_f32 = (float*)alloc((size_t)LAYERS * NN * CH * 4);
  f16*   xs_f16 = (f16*)  alloc((size_t)LAYERS * NN * CH * 2);
  f16*   x0_f16 = (f16*)  alloc((size_t)NN * CH * 2);
  f16*   preT   = (f16*)  alloc((size_t)LAYERS * 192 * 64 * 2);
  f16*   postT  = (f16*)  alloc((size_t)LAYERS * 576 * 64 * 2);
  f16*   linT   = (f16*)  alloc((size_t)LAYERS * 64 * 64 * 2);
  f16*   asapT  = (f16*)  alloc((size_t)64 * 64 * 2);
  f16*   wihTf  = (f16*)  alloc((size_t)64 * 768 * 2);
  f16*   whhTf  = (f16*)  alloc((size_t)192 * 768 * 2);
  f16*   wihTb  = (f16*)  alloc((size_t)64 * 768 * 2);
  f16*   whhTb  = (f16*)  alloc((size_t)192 * 768 * 2);
  float* xcmb   = (float*)alloc((size_t)NN * CH * 4);
  const size_t scratchBase = off;
  // phase 1 (per PNA layer)
  size_t o1 = scratchBase;
  auto a1 = [&](size_t b) -> char* { char* p = ws + o1; o1 = alignUp(o1 + b); return p; };
  f16*   efb  = (f16*)  a1((size_t)NE * CH * 2);
  float* mbuf = (float*)a1((size_t)NE * CH * 4);
  float* s1b  = (float*)a1((size_t)NN * CH * 4);
  float* s2b  = (float*)a1((size_t)NN * CH * 4);
  float* mxb  = (float*)a1((size_t)NN * CH * 4);
  float* mnb  = (float*)a1((size_t)NN * CH * 4);
  float* degb = (float*)a1((size_t)NN * 4);
  f16*   out9 = (f16*)  a1((size_t)NN * 576 * 2);
  f16*   tmp16= (f16*)  a1((size_t)NN * CH * 2);
  // phase 2 (LSTM + ASAP + MLP) -- reuses phase-1 scratch
  size_t o2 = scratchBase;
  auto a2 = [&](size_t b) -> char* { char* p = ws + o2; o2 = alignUp(o2 + b); return p; };
  f16*   hA    = (f16*)  a2((size_t)NN * HLSTM * 2);
  f16*   hB    = (f16*)  a2((size_t)NN * HLSTM * 2);
  float* cSt   = (float*)a2((size_t)NN * HLSTM * 4);
  float* alpha = (float*)a2((size_t)LAYERS * NN * 4);
  float* maxb  = (float*)a2((size_t)NN * CH * 4);
  f16*   maxb16= (f16*)  a2((size_t)NN * CH * 2);
  float* xq    = (float*)a2((size_t)NN * CH * 4);
  float* sArr  = (float*)a2((size_t)E2T * 4);
  float* esArr = (float*)a2((size_t)E2T * 4);
  float* smax  = (float*)a2((size_t)NN * 4);
  float* ssum  = (float*)a2((size_t)NN * 4);
  float* vbuf  = (float*)a2((size_t)NN * CH * 4);
  float* aN    = (float*)a2((size_t)NN * 4);
  float* bN    = (float*)a2((size_t)NN * 4);
  float* fitb  = (float*)a2((size_t)NN * 4);
  float* fitn  = (float*)a2((size_t)NN * 4);
  float* mlpin = (float*)a2((size_t)NG * 1024 * 4);
  float* mlpA  = (float*)a2((size_t)NG * 512 * 4);
  float* mlpB  = (float*)a2((size_t)NG * 512 * 4);

  const dim3 B256(256), B64(64);
  const dim3 GB(32, 4);  // GEMM blocks: 4 waves

  // -------- weight conversions (f32 -> fragment-packed f16)
  for (int i = 0; i < LAYERS; ++i) {
    k_wFrag<<<gdiv(192 * 64, 256), B256, 0, stream>>>(preT + (size_t)i * 192 * 64, pre_w + (size_t)i * 64 * 192, 64, 192);
    k_wFrag<<<gdiv(576 * 64, 256), B256, 0, stream>>>(postT + (size_t)i * 576 * 64, post_w + (size_t)i * 64 * 576, 64, 576);
    k_wFrag<<<gdiv(64 * 64, 256), B256, 0, stream>>>(linT + (size_t)i * 64 * 64, lin_w + (size_t)i * 64 * 64, 64, 64);
  }
  k_wFrag<<<gdiv(64 * 64, 256), B256, 0, stream>>>(asapT, asLinW, 64, 64);
  k_wFrag<<<gdiv(64 * 768, 256), B256, 0, stream>>>(wihTf, wihF, 768, 64);
  k_wFrag<<<gdiv(192 * 768, 256), B256, 0, stream>>>(whhTf, whhF, 768, 192);
  k_wFrag<<<gdiv(64 * 768, 256), B256, 0, stream>>>(wihTb, wihB, 768, 64);
  k_wFrag<<<gdiv(192 * 768, 256), B256, 0, stream>>>(whhTb, whhB, 768, 192);
  k_f32_to_f16<<<gdiv((long)NN * CH, 256), B256, 0, stream>>>(x0_f16, x_in, NN * CH);

  // -------- PNA layers
  const float* xcurF = x_in;
  const f16*   xcurH = x0_f16;
  for (int i = 0; i < LAYERS; ++i) {
    k_edge_e<<<gdiv((long)NE * CH, 256), B256, 0, stream>>>(
        ea_in, edge_w + (size_t)i * 64 * 8, edge_b + (size_t)i * 64, efb);
    k_msg_gemm<<<NE / 16, GB, 0, stream>>>(
        xcurH, srcA, dstA, efb, preT + (size_t)i * 192 * 64, pre_b + (size_t)i * 64, mbuf);
    k_zero_u32<<<gdiv((long)NN * CH, 256), B256, 0, stream>>>((unsigned*)s1b, NN * CH);
    k_zero_u32<<<gdiv((long)NN * CH, 256), B256, 0, stream>>>((unsigned*)s2b, NN * CH);
    k_fill_f32<<<gdiv((long)NN * CH, 256), B256, 0, stream>>>(mxb, -1e30f, NN * CH);
    k_fill_f32<<<gdiv((long)NN * CH, 256), B256, 0, stream>>>(mnb, 1e30f, NN * CH);
    k_zero_u32<<<gdiv(NN, 256), B256, 0, stream>>>((unsigned*)degb, NN);
    k_scatter_agg<<<gdiv((long)NE * CH, 256), B256, 0, stream>>>(dstA, mbuf, s1b, s2b, mxb, mnb, degb);
    k_build_out9<<<gdiv((long)NN * CH, 256), B256, 0, stream>>>(xcurF, s1b, s2b, mxb, mnb, degb, adl + i, out9);
    k_gemm64<576><<<NN / 16, GB, 0, stream>>>(out9, postT + (size_t)i * 576 * 64,
                                              post_b + (size_t)i * 64, (float*)nullptr, tmp16);
    k_gemm64<64><<<NN / 16, GB, 0, stream>>>(tmp16, linT + (size_t)i * 64 * 64,
                                             lin_b + (size_t)i * 64,
                                             xs_f32 + (size_t)i * NN * CH, xs_f16 + (size_t)i * NN * CH);
    xcurF = xs_f32 + (size_t)i * NN * CH;
    xcurH = xs_f16 + (size_t)i * NN * CH;
  }

  // -------- JK bi-LSTM (attention logits only; no h-history materialized)
  k_zero_u32<<<gdiv((long)LAYERS * NN, 256), B256, 0, stream>>>((unsigned*)alpha, LAYERS * NN);
  for (int dir = 0; dir < 2; ++dir) {
    k_zero_u32<<<gdiv((long)NN * HLSTM / 2, 256), B256, 0, stream>>>((unsigned*)hA, NN * HLSTM / 2);
    k_zero_u32<<<gdiv((long)NN * HLSTM / 2, 256), B256, 0, stream>>>((unsigned*)hB, NN * HLSTM / 2);
    k_zero_u32<<<gdiv((long)NN * HLSTM, 256), B256, 0, stream>>>((unsigned*)cSt, NN * HLSTM);
    f16* hR = hA; f16* hW = hB;
    const f16* wi = dir == 0 ? wihTf : wihTb;
    const f16* wh = dir == 0 ? whhTf : whhTb;
    const float* bi = dir == 0 ? bihF : bihB;
    const float* bh = dir == 0 ? bhhF : bhhB;
    const int attOff = dir == 0 ? 0 : HLSTM;
    for (int s = 0; s < LAYERS; ++s) {
      int t = dir == 0 ? s : (LAYERS - 1 - s);
      k_lstm_step<<<dim3(NN / 16, HLSTM / 16), GB, 0, stream>>>(
          xs_f16 + (size_t)t * NN * CH, hR, hW, cSt, wi, wh, bi, bh, jkAttW, attOff, alpha + (size_t)t * NN);
      f16* tmp = hR; hR = hW; hW = tmp;
    }
  }
  k_jk_combine<<<gdiv((long)NN * CH, 256), B256, 0, stream>>>(alpha, jkAttB, xs_f32, xcmb);

  // -------- ASAP pooling
  k_copy_f32<<<gdiv((long)NN * CH, 256), B256, 0, stream>>>(maxb, xcmb, NN * CH);  // self-loop seed
  k_scatter_max_x<<<gdiv((long)NE * CH, 256), B256, 0, stream>>>(srcA, dstA, xcmb, maxb);
  k_f32_to_f16<<<gdiv((long)NN * CH, 256), B256, 0, stream>>>(maxb16, maxb, NN * CH);
  k_gemm64<64><<<NN / 16, GB, 0, stream>>>(maxb16, asapT, asLinB, xq, (f16*)nullptr);
  k_asap_score<<<gdiv(E2T, 256), B256, 0, stream>>>(srcA, dstA, xq, xcmb, asAttW, asAttB, sArr);
  k_fill_f32<<<gdiv(NN, 256), B256, 0, stream>>>(smax, -1e30f, NN);
  k_scatter_smax<<<gdiv(E2T, 256), B256, 0, stream>>>(dstA, sArr, smax);
  k_zero_u32<<<gdiv(NN, 256), B256, 0, stream>>>((unsigned*)ssum, NN);
  k_seg_exp<<<gdiv(E2T, 256), B256, 0, stream>>>(dstA, sArr, smax, esArr, ssum);
  k_zero_u32<<<gdiv((long)NN * CH, 256), B256, 0, stream>>>((unsigned*)vbuf, NN * CH);
  k_scatter_v<<<gdiv((long)E2T * CH, 256), B256, 0, stream>>>(srcA, dstA, xcmb, esArr, ssum, vbuf);
  k_leconv_node<<<gdiv(NN, 256), B256, 0, stream>>>(vbuf, leW1, leB1, leW2, leW3, leB3, aN, bN, fitb);
  k_scatter_fit<<<gdiv(E2T, 256), B256, 0, stream>>>(srcA, dstA, aN, bN, fitb);
  k_sigmoid<<<gdiv(NN, 256), B256, 0, stream>>>(fitb, fitn, NN);
  k_topk_gather<<<NG, B256, 0, stream>>>(fitn, vbuf, mlpin);

  // -------- MLP head: 1024->512->...->2 (GELU) then 2->1
  static const int dims[10] = {1024, 512, 256, 128, 64, 32, 16, 8, 4, 2};
  const float* cur = mlpin;
  float* nxt = mlpA;
  for (int i = 0; i < 9; ++i) {
    const float* w = (const float*)d_in[32 + 2 * i];
    const float* b = (const float*)d_in[33 + 2 * i];
    k_dense<<<gdiv((long)NG * dims[i + 1], 64), B64, 0, stream>>>(cur, w, b, nxt, dims[i], dims[i + 1], 1);
    cur = nxt;
    nxt = (nxt == mlpA) ? mlpB : mlpA;
  }
  k_dense<<<1, B64, 0, stream>>>(cur, (const float*)d_in[50], (const float*)d_in[51], out, 2, 1, 0);
}
